// KNNGNN_16999480557903
// MI455X (gfx1250) — compile-verified, matrix operand order checked by reading, not run on hardware
//
#include <hip/hip_runtime.h>
#include <hip/hip_bf16.h>

// ---------------------------------------------------------------------------
// Problem constants (match reference setup_inputs)
// ---------------------------------------------------------------------------
#define ND   4096
#define NM   8192
#define NNODE (ND + NM)      // 12288
#define NEDGE 262144
#define KNN  50

typedef __attribute__((ext_vector_type(16))) _Float16 v16h;
typedef __attribute__((ext_vector_type(8)))  _Float16 v8h;
typedef __attribute__((ext_vector_type(8)))  float    v8f;

// ---------------------------------------------------------------------------
// LDS-staged WMMA GEMM:  C[M x Nc] = act( A[M x K] @ B[K x Nc] + bias )
// Row-major global layout. Block = 256 threads = 8 waves arranged 2(M) x 4(N):
// workgroup tile 32(M) x 64(N), K-step 64 (2 x v_wmma_f32_16x16x32_f16).
// A tile and B tile are cooperatively staged once per block into LDS as f16
// (B stored transposed) so every WMMA fragment is a contiguous 16-byte
// ds_load_b128. Requirements (all call sites satisfy them):
//   M % 32 == 0, Nc % 64 == 0, K % 64 == 0, grid exactly tiled.
// act: 0 = none, 1 = relu, 2 = elu
// ---------------------------------------------------------------------------
__global__ __launch_bounds__(256)
void gemm_wmma_f16(const float* __restrict__ A, const float* __restrict__ B,
                   const float* __restrict__ bias, float* __restrict__ C,
                   int M, int Nc, int K, int lda, int ldb, int ldc, int act)
{
    __shared__ _Float16 sA[32 * 64];    // [row][k]   4 KB
    __shared__ _Float16 sBt[64 * 64];   // [col][k]   8 KB  (transposed)

    const int tid  = threadIdx.x;
    const int wave = tid >> 5;
    const int lane = tid & 31;
    const int wm   = wave >> 2;          // 0..1
    const int wn   = wave & 3;           // 0..3
    const int mBase = blockIdx.y * 32;
    const int nBase = blockIdx.x * 64;

    const int r15   = lane & 15;
    const int klane = (lane < 16) ? 0 : 8;   // fragment K sub-offset
    const int arow_l = wm * 16 + r15;        // local A row   (0..31)
    const int bcol_l = wn * 16 + r15;        // local B col   (0..63)

    // cooperative-load coordinates
    const int la_r = tid >> 3;               // A: row 0..31
    const int la_k = (tid & 7) * 8;          // A: k   0..56
    const int lb_k = tid >> 2;               // B: k   0..63
    const int lb_c = (tid & 3) * 16;         // B: col 0,16,32,48

    v8f acc = {};
    for (int kb = 0; kb < K; kb += 64) {
        // ---- stage A tile (32x64 f32 -> f16), one b128 LDS store/thread ----
        {
            const float* ap = A + (size_t)(mBase + la_r) * lda + kb + la_k;
            float4 f0 = *(const float4*)ap;
            float4 f1 = *(const float4*)(ap + 4);
            v8h hv;
            hv[0] = (_Float16)f0.x; hv[1] = (_Float16)f0.y;
            hv[2] = (_Float16)f0.z; hv[3] = (_Float16)f0.w;
            hv[4] = (_Float16)f1.x; hv[5] = (_Float16)f1.y;
            hv[6] = (_Float16)f1.z; hv[7] = (_Float16)f1.w;
            *(v8h*)&sA[la_r * 64 + la_k] = hv;
        }
        // ---- stage B tile (64x64 f32 -> f16, transposed into sBt) ----
        {
            const float* bp = B + (size_t)(kb + lb_k) * ldb + nBase + lb_c;
            const float4* bq = (const float4*)bp;
#pragma unroll
            for (int jj = 0; jj < 4; ++jj) {
                float4 q = bq[jj];
                sBt[(lb_c + jj * 4 + 0) * 64 + lb_k] = (_Float16)q.x;
                sBt[(lb_c + jj * 4 + 1) * 64 + lb_k] = (_Float16)q.y;
                sBt[(lb_c + jj * 4 + 2) * 64 + lb_k] = (_Float16)q.z;
                sBt[(lb_c + jj * 4 + 3) * 64 + lb_k] = (_Float16)q.w;
            }
        }
        // prefetch next K-tile while this one is consumed (global_prefetch_b8)
        if (kb + 64 < K) {
            __builtin_prefetch(A + (size_t)(mBase + la_r) * lda + kb + 64 + la_k, 0, 1);
            __builtin_prefetch(B + (size_t)(kb + 64 + lb_k) * ldb + nBase + lb_c, 0, 1);
        }
        __syncthreads();

        // ---- compute: 2 WMMAs over the 64-wide K tile ----
#pragma unroll
        for (int kk = 0; kk < 64; kk += 32) {
            v8h a0 = *(const v8h*)&sA[arow_l * 64 + kk + klane];
            v8h a1 = *(const v8h*)&sA[arow_l * 64 + kk + klane + 16];
            v8h b0 = *(const v8h*)&sBt[bcol_l * 64 + kk + klane];
            v8h b1 = *(const v8h*)&sBt[bcol_l * 64 + kk + klane + 16];
            v16h a = __builtin_shufflevector(a0, a1, 0, 1, 2, 3, 4, 5, 6, 7,
                                             8, 9, 10, 11, 12, 13, 14, 15);
            v16h b = __builtin_shufflevector(b0, b1, 0, 1, 2, 3, 4, 5, 6, 7,
                                             8, 9, 10, 11, 12, 13, 14, 15);
            acc = __builtin_amdgcn_wmma_f32_16x16x32_f16(
                      false, a, false, b, (short)0, acc, false, false);
        }
        __syncthreads();
    }

    // ---- epilogue: bias + activation, direct global store ----
    const int bcol = nBase + bcol_l;
#pragma unroll
    for (int j = 0; j < 8; ++j) {
        const int r = mBase + wm * 16 + j + 8 * (lane >> 4);
        float v = acc[j];
        if (bias) v += bias[bcol];
        if (act == 1)      v = v > 0.f ? v : 0.f;
        else if (act == 2) v = v > 0.f ? v : (__expf(v) - 1.f);
        C[(size_t)r * ldc + bcol] = v;
    }
}

// ---------------------------------------------------------------------------
// Helpers / small kernels
// ---------------------------------------------------------------------------
__device__ __forceinline__ float atomicMaxFloat(float* addr, float v)
{
    return (v >= 0.f)
        ? __int_as_float(atomicMax((int*)addr, __float_as_int(v)))
        : __uint_as_float(atomicMin((unsigned int*)addr, __float_as_uint(v)));
}

__global__ void fill_val(float* p, float v, int n)
{
    int i = blockIdx.x * blockDim.x + threadIdx.x;
    if (i < n) p[i] = v;
}

// asrc[n,h] = <hx[n, h*C : (h+1)*C], att_s[h]> ; adst likewise. One warp per (n,h).
__global__ void att_dots(const float* __restrict__ hx,
                         const float* __restrict__ att_s,
                         const float* __restrict__ att_d,
                         float* __restrict__ asrc, float* __restrict__ adst,
                         int Nn, int H, int C)
{
    int warp = (blockIdx.x * blockDim.x + threadIdx.x) >> 5;
    int lane = threadIdx.x & 31;
    if (warp >= Nn * H) return;
    int n = warp / H, h = warp % H;
    const float* v = hx + (size_t)n * H * C + (size_t)h * C;
    float ss = 0.f, sd = 0.f;
    for (int c = lane; c < C; c += 32) {
        float x = v[c];
        ss += x * att_s[h * C + c];
        sd += x * att_d[h * C + c];
    }
    for (int o = 16; o > 0; o >>= 1) {
        ss += __shfl_xor(ss, o, 32);
        sd += __shfl_xor(sd, o, 32);
    }
    if (lane == 0) { asrc[n * H + h] = ss; adst[n * H + h] = sd; }
}

// e = leaky_relu(asrc[src]+adst[dst], 0.2); segment max over dst.
__global__ void edge_max(const int* __restrict__ src, const int* __restrict__ dst,
                         const float* __restrict__ asrc, const float* __restrict__ adst,
                         float* __restrict__ ebuf, float* __restrict__ segmax,
                         int Ne, int H)
{
    int i = blockIdx.x * blockDim.x + threadIdx.x;
    if (i >= Ne * H) return;
    int e = i / H, h = i % H;
    int s = src[e], d = dst[e];
    float v = asrc[s * H + h] + adst[d * H + h];
    v = (v >= 0.f) ? v : 0.2f * v;
    ebuf[i] = v;
    atomicMaxFloat(&segmax[d * H + h], v);
}

// ex = exp(e - m[dst]) (m<-0 if not finite); segment sum of ex over dst.
__global__ void edge_expsum(const int* __restrict__ dst,
                            float* __restrict__ ebuf,
                            const float* __restrict__ segmax,
                            float* __restrict__ segsum, int Ne, int H)
{
    int i = blockIdx.x * blockDim.x + threadIdx.x;
    if (i >= Ne * H) return;
    int e = i / H, h = i % H;
    int d = dst[e];
    float m = segmax[d * H + h];
    if (!__builtin_isfinite(m)) m = 0.f;
    float ex = __expf(ebuf[i] - m);
    ebuf[i] = ex;
    atomicAdd(&segsum[d * H + h], ex);
}

// out[dst] += alpha * hx[src]; one warp per edge, all heads.
__global__ void edge_aggregate(const int* __restrict__ src, const int* __restrict__ dst,
                               const float* __restrict__ hx,
                               const float* __restrict__ ebuf,
                               const float* __restrict__ segsum,
                               float* __restrict__ out, int Ne, int H, int C)
{
    int warp = (blockIdx.x * blockDim.x + threadIdx.x) >> 5;
    int lane = threadIdx.x & 31;
    if (warp >= Ne) return;
    int s = src[warp], d = dst[warp];
    for (int h = 0; h < H; ++h) {
        float den = segsum[d * H + h];
        den = den > 1e-16f ? den : 1e-16f;
        float alpha = ebuf[(size_t)warp * H + h] / den;
        const float* v = hx + (size_t)s * H * C + (size_t)h * C;
        float* o = out + (size_t)d * H * C + (size_t)h * C;
        for (int c = lane; c < C; c += 32)
            atomicAdd(&o[c], alpha * v[c]);
    }
}

// out[r, c] = act(in[r*cols+c] + bias[c])   (out has leading dim ldo)
__global__ void add_bias_store(const float* __restrict__ in,
                               const float* __restrict__ bias,
                               float* __restrict__ out,
                               int n, int cols, int ldo, int act)
{
    int i = blockIdx.x * blockDim.x + threadIdx.x;
    if (i >= n * cols) return;
    int r = i / cols, c = i % cols;
    float v = in[i] + bias[c];
    if (act == 1) v = v > 0.f ? v : 0.f;
    out[(size_t)r * ldo + c] = v;
}

// g[n, 0:4] = fs[n, 0:512] @ W_proj + b_proj ; one warp per node.
__global__ void proj_kernel(const float* __restrict__ fs,
                            const float* __restrict__ W, const float* __restrict__ b,
                            float* __restrict__ g, int Nn)
{
    int warp = (blockIdx.x * blockDim.x + threadIdx.x) >> 5;
    int lane = threadIdx.x & 31;
    if (warp >= Nn) return;
    const float* row = fs + (size_t)warp * 1024;
    float a0 = 0.f, a1 = 0.f, a2 = 0.f, a3 = 0.f;
    for (int c = lane; c < 512; c += 32) {
        float x = row[c];
        a0 += x * W[c * 4 + 0]; a1 += x * W[c * 4 + 1];
        a2 += x * W[c * 4 + 2]; a3 += x * W[c * 4 + 3];
    }
    for (int o = 16; o > 0; o >>= 1) {
        a0 += __shfl_xor(a0, o, 32); a1 += __shfl_xor(a1, o, 32);
        a2 += __shfl_xor(a2, o, 32); a3 += __shfl_xor(a3, o, 32);
    }
    if (lane == 0) {
        g[warp * 4 + 0] = a0 + b[0]; g[warp * 4 + 1] = a1 + b[1];
        g[warp * 4 + 2] = a2 + b[2]; g[warp * 4 + 3] = a3 + b[3];
    }
}

// Exact k-NN (k=50, includes self) in 4-D; one thread per query node.
// Insertion-sorted list; ties keep the earlier index (matches top_k).
__global__ void knn_topk(const float* __restrict__ g, int* __restrict__ idx, int Nn)
{
    int n = blockIdx.x * blockDim.x + threadIdx.x;
    if (n >= Nn) return;
    float gx = g[n * 4 + 0], gy = g[n * 4 + 1];
    float gz = g[n * 4 + 2], gw = g[n * 4 + 3];
    float bd[KNN]; int bi[KNN];
#pragma unroll
    for (int k = 0; k < KNN; ++k) { bd[k] = __builtin_inff(); bi[k] = 0; }
    for (int j = 0; j < Nn; ++j) {
        float dx = gx - g[j * 4 + 0], dy = gy - g[j * 4 + 1];
        float dz = gz - g[j * 4 + 2], dw = gw - g[j * 4 + 3];
        float d = dx * dx + dy * dy + dz * dz + dw * dw;
        if (d < bd[KNN - 1]) {
            int p = KNN - 1;
            while (p > 0 && bd[p - 1] > d) { bd[p] = bd[p - 1]; bi[p] = bi[p - 1]; --p; }
            bd[p] = d; bi[p] = j;
        }
    }
#pragma unroll
    for (int k = 0; k < KNN; ++k) idx[(size_t)n * KNN + k] = bi[k];
}

// fs[n, 512+c] = mean_k fs[idx[n,k], c]   (c in [0,512)); one block per node.
__global__ void knn_mean(float* __restrict__ fs, const int* __restrict__ idx, int Nn)
{
    int n = blockIdx.x;
    int c = threadIdx.x;          // blockDim = 512
    float s = 0.f;
    for (int k = 0; k < KNN; ++k)
        s += fs[(size_t)idx[(size_t)n * KNN + k] * 1024 + c];
    fs[(size_t)n * 1024 + 512 + c] = s * (1.f / KNN);
}

// In-place LayerNorm over fs[:, 512:1024]; one block (512 thr, 16 waves)/node.
__global__ void layernorm(float* __restrict__ fs,
                          const float* __restrict__ gam, const float* __restrict__ bet,
                          int Nn)
{
    __shared__ float red[16];
    __shared__ float mu_s, rstd_s;
    int n = blockIdx.x, t = threadIdx.x;
    int lane = t & 31, w = t >> 5;
    float v = fs[(size_t)n * 1024 + 512 + t];

    float s = v;
    for (int o = 16; o > 0; o >>= 1) s += __shfl_xor(s, o, 32);
    if (lane == 0) red[w] = s;
    __syncthreads();
    if (t == 0) {
        float m = 0.f;
        for (int i = 0; i < 16; ++i) m += red[i];
        mu_s = m * (1.f / 512.f);
    }
    __syncthreads();
    float mu = mu_s;
    float d = v - mu, s2 = d * d;
    for (int o = 16; o > 0; o >>= 1) s2 += __shfl_xor(s2, o, 32);
    if (lane == 0) red[w] = s2;
    __syncthreads();
    if (t == 0) {
        float m = 0.f;
        for (int i = 0; i < 16; ++i) m += red[i];
        rstd_s = rsqrtf(m * (1.f / 512.f) + 1e-5f);
    }
    __syncthreads();
    fs[(size_t)n * 1024 + 512 + t] = d * rstd_s * gam[t] + bet[t];
}

// hf[:, 64:128] = feats
__global__ void copy_feats(const float* __restrict__ feats, float* __restrict__ hf, int Nn)
{
    int i = blockIdx.x * blockDim.x + threadIdx.x;
    if (i >= Nn * 64) return;
    int r = i >> 6, c = i & 63;
    hf[(size_t)r * 128 + 64 + c] = feats[i];
}

// out[e] = sigmoid( [hh[dis[e]], hh[mir[e]]] @ W_pred + b_pred )
__global__ void pred_kernel(const float* __restrict__ hh,
                            const int* __restrict__ dis, const int* __restrict__ mir,
                            const float* __restrict__ Wp, const float* __restrict__ bp,
                            float* __restrict__ out, int Ne)
{
    int e = blockIdx.x * blockDim.x + threadIdx.x;
    if (e >= Ne) return;
    const float* a = hh + (size_t)dis[e] * 64;
    const float* b = hh + (size_t)mir[e] * 64;
    float s = bp[0];
#pragma unroll 8
    for (int c = 0; c < 64; ++c) s += a[c] * Wp[c] + b[c] * Wp[64 + c];
    out[e] = 1.f / (1.f + __expf(-s));
}

// ---------------------------------------------------------------------------
// Orchestration
// ---------------------------------------------------------------------------
extern "C" void kernel_launch(void* const* d_in, const int* in_sizes, int n_in,
                              void* d_out, int out_size, void* d_ws, size_t ws_size,
                              hipStream_t stream)
{
    (void)in_sizes; (void)n_in; (void)out_size; (void)ws_size;
    const float* d_sim    = (const float*)d_in[0];
    const float* m_sim    = (const float*)d_in[1];
    const int*   diseases = (const int*)  d_in[2];
    const int*   mirnas   = (const int*)  d_in[3];
    const float* W_dfc    = (const float*)d_in[4];
    const float* W_mfc    = (const float*)d_in[5];
    const float* W_lin1   = (const float*)d_in[6];
    const float* b_lin1   = (const float*)d_in[7];
    const float* W_gat1   = (const float*)d_in[8];
    const float* att_src1 = (const float*)d_in[9];
    const float* att_dst1 = (const float*)d_in[10];
    const float* b_gat1   = (const float*)d_in[11];
    const float* W_gat2   = (const float*)d_in[12];
    const float* att_src2 = (const float*)d_in[13];
    const float* att_dst2 = (const float*)d_in[14];
    const float* b_gat2   = (const float*)d_in[15];
    const float* W_proj   = (const float*)d_in[16];
    const float* b_proj   = (const float*)d_in[17];
    const float* ln_g     = (const float*)d_in[18];
    const float* ln_b     = (const float*)d_in[19];
    const float* W_lin    = (const float*)d_in[20];
    const float* b_lin    = (const float*)d_in[21];
    const float* W_dfc1   = (const float*)d_in[22];
    const float* b_dfc1   = (const float*)d_in[23];
    const float* W_mfc1   = (const float*)d_in[24];
    const float* b_mfc1   = (const float*)d_in[25];
    const float* W_pred   = (const float*)d_in[26];
    const float* b_pred   = (const float*)d_in[27];
    float* out = (float*)d_out;

    const int Nn = NNODE, Ne = NEDGE;

    // ---- workspace layout (floats), ~303 MB total ----
    float* w = (float*)d_ws;
    size_t off = 0;
    auto alloc = [&](size_t n) { float* p = w + off; off += n; return p; };
    float* feats   = alloc((size_t)Nn * 64);
    float* fs      = alloc((size_t)Nn * 1024);   // [x1 | x2 | sim]
    float* h1      = alloc((size_t)Nn * 2048);   // GAT1 transformed features
    float* g1      = alloc((size_t)Nn * 2048);   // GAT1 aggregate -> relu in place
    float* asrc1   = alloc((size_t)Nn * 8);
    float* adst1   = alloc((size_t)Nn * 8);
    float* ebuf1   = alloc((size_t)Ne * 8);
    float* segmax1 = alloc((size_t)Nn * 8);
    float* segsum1 = alloc((size_t)Nn * 8);
    float* h2      = alloc((size_t)Nn * 256);
    float* asrc2   = alloc((size_t)Nn);
    float* adst2   = alloc((size_t)Nn);
    float* ebuf2   = alloc((size_t)Ne);
    float* segmax2 = alloc((size_t)Nn);
    float* segsum2 = alloc((size_t)Nn);
    float* agg2    = alloc((size_t)Nn * 256);
    float* g4      = alloc((size_t)Nn * 4);
    float* hf      = alloc((size_t)Nn * 128);    // [h | feats]
    float* hh      = alloc((size_t)Nn * 64);
    int*   knn     = (int*)(w + off); off += (size_t)Nn * KNN;

    const dim3 B256(256);

    // 1) feats = [d_sim @ W_dfc ; m_sim @ W_mfc]
    gemm_wmma_f16<<<dim3(64 / 64, ND / 32), B256, 0, stream>>>(
        d_sim, W_dfc, nullptr, feats, ND, 64, ND, ND, 64, 64, 0);
    gemm_wmma_f16<<<dim3(64 / 64, NM / 32), B256, 0, stream>>>(
        m_sim, W_mfc, nullptr, feats + (size_t)ND * 64, NM, 64, NM, NM, 64, 64, 0);

    // 2) x1 = relu(feats @ W_lin1 + b_lin1)  -> fs[:, 0:256]
    gemm_wmma_f16<<<dim3(256 / 64, Nn / 32), B256, 0, stream>>>(
        feats, W_lin1, b_lin1, fs, Nn, 256, 64, 64, 256, 1024, 1);

    // 3) h1 = feats @ W_gat1 (bias applied after aggregation)
    gemm_wmma_f16<<<dim3(2048 / 64, Nn / 32), B256, 0, stream>>>(
        feats, W_gat1, nullptr, h1, Nn, 2048, 64, 64, 2048, 2048, 0);

    // 4) GAT layer 1 (H=8, C=256)
    att_dots<<<(Nn * 8 * 32 + 255) / 256, B256, 0, stream>>>(
        h1, att_src1, att_dst1, asrc1, adst1, Nn, 8, 256);
    fill_val<<<(Nn * 8 + 255) / 256, B256, 0, stream>>>(segmax1, -__builtin_inff(), Nn * 8);
    hipMemsetAsync(segsum1, 0, (size_t)Nn * 8 * sizeof(float), stream);
    hipMemsetAsync(g1, 0, (size_t)Nn * 2048 * sizeof(float), stream);
    edge_max<<<(Ne * 8 + 255) / 256, B256, 0, stream>>>(
        diseases, mirnas, asrc1, adst1, ebuf1, segmax1, Ne, 8);
    edge_expsum<<<(Ne * 8 + 255) / 256, B256, 0, stream>>>(
        mirnas, ebuf1, segmax1, segsum1, Ne, 8);
    edge_aggregate<<<(Ne * 32 + 255) / 256, B256, 0, stream>>>(
        diseases, mirnas, h1, ebuf1, segsum1, g1, Ne, 8, 256);
    add_bias_store<<<((size_t)Nn * 2048 + 255) / 256, B256, 0, stream>>>(
        g1, b_gat1, g1, Nn, 2048, 2048, 1);   // g1 = relu(agg + b_gat1), in place

    // 5) h2 = g1 @ W_gat2
    gemm_wmma_f16<<<dim3(256 / 64, Nn / 32), B256, 0, stream>>>(
        g1, W_gat2, nullptr, h2, Nn, 256, 2048, 2048, 256, 256, 0);

    // 6) GAT layer 2 (H=1, C=256) -> x2 -> fs[:, 256:512]
    att_dots<<<(Nn * 32 + 255) / 256, B256, 0, stream>>>(
        h2, att_src2, att_dst2, asrc2, adst2, Nn, 1, 256);
    fill_val<<<(Nn + 255) / 256, B256, 0, stream>>>(segmax2, -__builtin_inff(), Nn);
    hipMemsetAsync(segsum2, 0, (size_t)Nn * sizeof(float), stream);
    hipMemsetAsync(agg2, 0, (size_t)Nn * 256 * sizeof(float), stream);
    edge_max<<<(Ne + 255) / 256, B256, 0, stream>>>(
        diseases, mirnas, asrc2, adst2, ebuf2, segmax2, Ne, 1);
    edge_expsum<<<(Ne + 255) / 256, B256, 0, stream>>>(
        mirnas, ebuf2, segmax2, segsum2, Ne, 1);
    edge_aggregate<<<(Ne * 32 + 255) / 256, B256, 0, stream>>>(
        diseases, mirnas, h2, ebuf2, segsum2, agg2, Ne, 1, 256);
    add_bias_store<<<((size_t)Nn * 256 + 255) / 256, B256, 0, stream>>>(
        agg2, b_gat2, fs + 256, Nn, 256, 1024, 0);

    // 7) g = final @ W_proj + b_proj (final = fs[:, 0:512])
    proj_kernel<<<(Nn * 32 + 255) / 256, B256, 0, stream>>>(fs, W_proj, b_proj, g4, Nn);

    // 8) exact KNN (k=50) + neighbor mean -> fs[:, 512:1024], then LayerNorm
    knn_topk<<<(Nn + 127) / 128, dim3(128), 0, stream>>>(g4, knn, Nn);
    knn_mean<<<Nn, dim3(512), 0, stream>>>(fs, knn, Nn);
    layernorm<<<Nn, dim3(512), 0, stream>>>(fs, ln_g, ln_b, Nn);

    // 9) h = [final, sim] @ W_lin + b_lin -> hf[:, 0:64]; hf[:, 64:128] = feats
    gemm_wmma_f16<<<dim3(64 / 64, Nn / 32), B256, 0, stream>>>(
        fs, W_lin, b_lin, hf, Nn, 64, 1024, 1024, 64, 128, 0);
    copy_feats<<<(Nn * 64 + 255) / 256, B256, 0, stream>>>(feats, hf, Nn);

    // 10) hh = elu([h|feats] @ {W_dfc1 (disease rows), W_mfc1 (mirna rows)} + b)
    gemm_wmma_f16<<<dim3(64 / 64, ND / 32), B256, 0, stream>>>(
        hf, W_dfc1, b_dfc1, hh, ND, 64, 128, 128, 64, 64, 2);
    gemm_wmma_f16<<<dim3(64 / 64, NM / 32), B256, 0, stream>>>(
        hf + (size_t)ND * 128, W_mfc1, b_mfc1, hh + (size_t)ND * 64, NM, 64, 128, 128, 64, 64, 2);

    // 11) edge prediction head + sigmoid
    pred_kernel<<<(Ne + 255) / 256, B256, 0, stream>>>(
        hh, diseases, mirnas, W_pred, b_pred, out, Ne);
}
